// PrunedModel_16733192585808
// MI455X (gfx1250) — matrix operand order repaired and checked by reading, not compiled, not run on hardware
//
#include <hip/hip_runtime.h>
#include <stdint.h>

// ---------------------------------------------------------------------------
// PrunedModel fused kernel for gfx1250 (MI455X)
//   LN(f32) -> gather -> bf16 WMMA GEMM (f32 acc) -> bias -> LDS scatter-add
// 512 threads (16 wave32) per WG, 32 rows per WG, software-pipelined B loads.
// ---------------------------------------------------------------------------

typedef __attribute__((ext_vector_type(16))) __bf16 v16bf;
typedef __attribute__((ext_vector_type(8)))  float  v8f;

#define ND     1024     // D (hidden dim / GEMM N)
#define KD     512      // P (pruned dim  / GEMM K)
#define ROWS   32       // rows (b,s) per workgroup
#define BLOCK  512      // 16 wave32
#define NWAVE  16

// LDS layout (bytes)
#define OFF_OUT   0                         // 32*1024*4 = 131072  residual tile f32
#define OFF_A     131072                    // 32*512*2  =  32768  gathered A bf16
#define OFF_NSCR  (OFF_A + 32768)           // 16*1024*2 =  32768  per-wave normed bf16
#define OFF_S3    (OFF_NSCR + 32768)        // 512*4     =   2048
#define OFF_S5    (OFF_S3 + 2048)           // 1024*4    =   4096
#define OFF_BS    (OFF_S5 + 4096)           // 1024*4    =   4096
#define SMEM_BYTES (OFF_BS + 4096)          // 206848

__device__ __forceinline__ unsigned short f2bf(float f) {
    union { float f; uint32_t u; } c; c.f = f;
    uint32_t u = c.u;
    uint32_t r = u + 0x7FFFu + ((u >> 16) & 1u);   // round-to-nearest-even
    return (unsigned short)(r >> 16);
}

// ---------------------------------------------------------------------------
// Pack W[n][k] (f32, n-major) into WMMA-B lane order, bf16.
// B tile (K=32 x N=16) for v_wmma_f32_16x16x32_bf16:
//   k order chosen here (A matches): lane = (k>=16)*16 + n ; 16 consecutive k per lane (8 VGPRs).
// Bpk[((n_tile*16 + k_tile)*32 + lane)*16 + j] = bf16(W[n][k_tile*32 + hi*16 + j])
// => each GEMM lane reads one contiguous 32B chunk per (tile, kstep).
// ---------------------------------------------------------------------------
__global__ void pack_w_kernel(const float* __restrict__ W,
                              unsigned short* __restrict__ Bpk) {
    int t    = blockIdx.x * blockDim.x + threadIdx.x;  // 0..32767
    int lane = t & 31;
    int kt   = (t >> 5) & 15;
    int nt   = t >> 9;                                  // 0..63
    int n    = nt * 16 + (lane & 15);
    int kb   = kt * 32 + (lane >> 4) * 16;
    const float* src = W + (size_t)n * KD + kb;
    unsigned short* dst = Bpk + (size_t)t * 16;
#pragma unroll
    for (int j = 0; j < 16; ++j) dst[j] = f2bf(src[j]);
}

__global__ __launch_bounds__(BLOCK)
void pruned_fused_kernel(const float* __restrict__ hidden,
                         const float* __restrict__ lnw,
                         const float* __restrict__ lnb,
                         const float* __restrict__ bias,
                         const int*   __restrict__ s3,
                         const int*   __restrict__ s5,
                         const unsigned short* __restrict__ Bpk,
                         float* __restrict__ out) {
    extern __shared__ char smem[];
    float*          out_lds = (float*)(smem + OFF_OUT);            // [32][1024]
    unsigned short* A_lds   = (unsigned short*)(smem + OFF_A);     // [32][512]
    unsigned short* nscr    = (unsigned short*)(smem + OFF_NSCR);  // [16][1024]
    int*            s3s     = (int*)(smem + OFF_S3);
    int*            s5s     = (int*)(smem + OFF_S5);
    float*          bs      = (float*)(smem + OFF_BS);

    const int tid  = threadIdx.x;
    const int lane = tid & 31;
    const int w    = tid >> 5;                // 0..15
    const int R0   = blockIdx.x * ROWS;

    // ---- stage 0: cache indices + bias ------------------------------------
    for (int i = tid; i < KD; i += BLOCK) s3s[i] = s3[i];
    for (int i = tid; i < ND; i += BLOCK) { s5s[i] = s5[i]; bs[i] = bias[i]; }
    __syncthreads();

    // ---- stage 1: LayerNorm + residual-seed + bf16 gather (2 rows/wave) ---
    const float4* lwp = (const float4*)lnw;
    const float4* lbp = (const float4*)lnb;
    for (int i = 0; i < 2; ++i) {
        const int rr = w * 2 + i;
        const size_t r = (size_t)(R0 + rr);
        const float4* hp = (const float4*)(hidden + r * ND);
        float4 x[8];
        float sum = 0.f, sq = 0.f;
#pragma unroll
        for (int q = 0; q < 8; ++q) {
            x[q] = hp[q * 32 + lane];
            sum += x[q].x + x[q].y + x[q].z + x[q].w;
            sq  += x[q].x*x[q].x + x[q].y*x[q].y + x[q].z*x[q].z + x[q].w*x[q].w;
        }
#pragma unroll
        for (int off = 16; off > 0; off >>= 1) {
            sum += __shfl_xor(sum, off, 32);
            sq  += __shfl_xor(sq,  off, 32);
        }
        const float mu   = sum * (1.f / ND);
        const float var  = sq * (1.f / ND) - mu * mu;
        const float rstd = rsqrtf(var + 1e-5f);

        float4* orow = (float4*)(out_lds + rr * ND);
        uint2*  nrow = (uint2*)(nscr + w * ND);
#pragma unroll
        for (int q = 0; q < 8; ++q) {
            const int c4 = q * 32 + lane;
            orow[c4] = x[q];                              // residual init (raw hidden)
            const float4 lw = lwp[c4], lb = lbp[c4];
            const float n0 = (x[q].x - mu) * rstd * lw.x + lb.x;
            const float n1 = (x[q].y - mu) * rstd * lw.y + lb.y;
            const float n2 = (x[q].z - mu) * rstd * lw.z + lb.z;
            const float n3 = (x[q].w - mu) * rstd * lw.w + lb.w;
            uint2 pk;
            pk.x = (uint32_t)f2bf(n0) | ((uint32_t)f2bf(n1) << 16);
            pk.y = (uint32_t)f2bf(n2) | ((uint32_t)f2bf(n3) << 16);
            nrow[c4] = pk;                                // 4 bf16 per store
        }
        // gather s3 channels for this row (same-wave LDS: in-order, no barrier)
#pragma unroll
        for (int j = 0; j < 16; ++j) {
            const int p = j * 32 + lane;
            A_lds[rr * KD + p] = nscr[w * ND + s3s[p]];
        }
    }
    __syncthreads();   // A-tile + residual tile complete across waves

    // ---- stage 2: GEMM  A[32x512]bf16 x B[512x1024]bf16 -> f32 ------------
    // wave -> m_tile = w&1 (16 rows), n-group = w>>1 (128 cols = 8 tiles)
    const int m_tile = w & 1;
    const int ngrp   = w >> 1;               // 0..7
    v8f acc[8] = {};

    const int arow  = m_tile * 16 + (lane & 15);
    const int ahalf = (lane >> 4) * 16;      // A: K = hi*16+{0..15}

    for (int kt = 0; kt < 16; ++kt) {
        // A fragment from LDS: two contiguous 16B runs of a row-major row
        union { v16bf v; uint4 q[2]; } af;
        const unsigned short* ap = A_lds + arow * KD + kt * 32 + ahalf;
        af.q[0] = *(const uint4*)(ap);
        af.q[1] = *(const uint4*)(ap + 8);

        // B fragments: 2-deep software pipeline over the 8 n-tiles.
        // tile t offset = t * 16*512 ushorts (= 16 KB)
        const unsigned short* bbase =
            Bpk + ((size_t)(ngrp * 8 * 16 + kt) * 512 + lane * 16);
        uint4 b0a = *(const uint4*)(bbase);
        uint4 b0b = *(const uint4*)(bbase + 8);
        uint4 b1a = *(const uint4*)(bbase + 8192);
        uint4 b1b = *(const uint4*)(bbase + 8192 + 8);
#pragma unroll
        for (int t = 0; t < 8; ++t) {
            uint4 na, nb;
            if (t + 2 < 8) {
                const unsigned short* p = bbase + (size_t)(t + 2) * 8192;
                na = *(const uint4*)(p);
                nb = *(const uint4*)(p + 8);
            }
            union { v16bf v; uint4 q[2]; } bu;
            bu.q[0] = b0a; bu.q[1] = b0b;
            acc[t] = __builtin_amdgcn_wmma_f32_16x16x32_bf16(
                false, af.v, false, bu.v, (short)0, acc[t], false, false);
            b0a = b1a; b0b = b1b;
            b1a = na;  b1b = nb;
        }
    }

    // ---- stage 3: bias + scatter-add into LDS residual tile ---------------
    // C layout: vgpr g, lane l -> row = m_tile*16 + g + (l>=16)*8, col = n_tile*16 + (l&15)
    const int rbase = m_tile * 16 + ((lane >> 4) << 3);
#pragma unroll
    for (int t = 0; t < 8; ++t) {
        const int col  = (ngrp * 8 + t) * 16 + (lane & 15);
        const int d    = s5s[col];
        const float bc = bs[col];
#pragma unroll
        for (int g = 0; g < 8; ++g) {
            atomicAdd(&out_lds[(rbase + g) * ND + d], acc[t][g] + bc);
        }
    }
    __syncthreads();

    // ---- stage 4: coalesced writeback -------------------------------------
    const float4* src = (const float4*)out_lds;
    float4* dst = (float4*)(out + (size_t)R0 * ND);
    for (int i = tid; i < ROWS * ND / 4; i += BLOCK) dst[i] = src[i];
}

extern "C" void kernel_launch(void* const* d_in, const int* in_sizes, int n_in,
                              void* d_out, int out_size, void* d_ws, size_t ws_size,
                              hipStream_t stream) {
    const float* hidden = (const float*)d_in[0];
    const float* lnw    = (const float*)d_in[1];
    const float* lnb    = (const float*)d_in[2];
    const float* W      = (const float*)d_in[3];   // [1024][512]
    const float* bias   = (const float*)d_in[4];
    const int*   s3     = (const int*)d_in[5];     // [512]
    const int*   s5     = (const int*)d_in[6];     // [1024]
    float*       out    = (float*)d_out;

    unsigned short* Bpk = (unsigned short*)d_ws;   // 1 MB packed bf16 W^T

    // repack W each call (deterministic; ~2 MB of traffic, negligible)
    pack_w_kernel<<<128, 256, 0, stream>>>(W, Bpk);

    (void)hipFuncSetAttribute((const void*)pruned_fused_kernel,
                              hipFuncAttributeMaxDynamicSharedMemorySize,
                              SMEM_BYTES);

    const int M = 8 * 4096;                        // B*S rows
    pruned_fused_kernel<<<M / ROWS, BLOCK, SMEM_BYTES, stream>>>(
        hidden, lnw, lnb, bias, s3, s5, Bpk, out);
}
